// MultiHeadAttention_83657372992140
// MI455X (gfx1250) — matrix operand order repaired and checked by reading, not compiled, not run on hardware
//
#include <hip/hip_runtime.h>

// ---------------------------------------------------------------------------
// MI455X (gfx1250) multi-head attention, bf16 WMMA pipeline:
//   0) convert inputs -> bf16, convert+transpose weights -> bf16 [n][k]
//   1) QKV projection GEMM: async-to-LDS double-buffered staging + WMMA
//   2) flash attention (online softmax), WMMA, fragments direct from global
//   3) output projection GEMM (same async WMMA body, fp32 out)
// ---------------------------------------------------------------------------

typedef __attribute__((ext_vector_type(16))) __bf16 bf16x16;
typedef __attribute__((ext_vector_type(8)))  __bf16 bf16x8;
typedef __attribute__((ext_vector_type(8)))  float  f32x8;

#define BATCH 2
#define SEQ   2048
#define DIM   1024
#define HEADS 16
#define HDIM  64
#define MROWS (BATCH * SEQ)   // 4096

static __device__ __forceinline__ bf16x16 mk16(bf16x8 lo, bf16x8 hi) {
    return __builtin_shufflevector(lo, hi, 0, 1, 2, 3, 4, 5, 6, 7,
                                           8, 9, 10, 11, 12, 13, 14, 15);
}

static __device__ __forceinline__ f32x8 wmma_bf16(bf16x16 a, bf16x16 b, f32x8 c) {
    return __builtin_amdgcn_wmma_f32_16x16x32_bf16(false, a, false, b,
                                                   (short)0, c, false, false);
}

// gfx1250 async global->LDS copy (ASYNCcnt tracked), 16B per lane
static __device__ __forceinline__ void async_b128(unsigned lds_off, const void* g) {
    asm volatile("global_load_async_to_lds_b128 %0, %1, off"
                 :: "v"(lds_off), "v"((unsigned long long)(uintptr_t)g)
                 : "memory");
}
static __device__ __forceinline__ void wait_async_4() {
    asm volatile("s_wait_asynccnt 0x4" ::: "memory");
}
static __device__ __forceinline__ void wait_async_0() {
    asm volatile("s_wait_asynccnt 0x0" ::: "memory");
}

// half-wave (16-lane) reductions; C-fragment rows live per 16-lane half
static __device__ __forceinline__ float halfmax16(float v) {
#pragma unroll
    for (int m = 1; m < 16; m <<= 1) v = fmaxf(v, __shfl_xor(v, m, 32));
    return v;
}
static __device__ __forceinline__ float halfsum16(float v) {
#pragma unroll
    for (int m = 1; m < 16; m <<= 1) v += __shfl_xor(v, m, 32);
    return v;
}

// ---------------------------------------------------------------------------
// Pre-pass: fp32 -> bf16 elementwise (q/k/v inputs)
// ---------------------------------------------------------------------------
__global__ __launch_bounds__(256)
void iconv_kernel(const float* __restrict__ q, const float* __restrict__ k,
                  const float* __restrict__ v,
                  __bf16* qb, __bf16* kb, __bf16* vb) {
    const float* in; __bf16* out;
    if (blockIdx.z == 0)      { in = q; out = qb; }
    else if (blockIdx.z == 1) { in = k; out = kb; }
    else                      { in = v; out = vb; }
    const size_t idx = ((size_t)blockIdx.x * 256 + threadIdx.x) * 8;
    float4 a = ((const float4*)(in + idx))[0];
    float4 b = ((const float4*)(in + idx))[1];
    bf16x8 t;
    t[0] = (__bf16)a.x; t[1] = (__bf16)a.y; t[2] = (__bf16)a.z; t[3] = (__bf16)a.w;
    t[4] = (__bf16)b.x; t[5] = (__bf16)b.y; t[6] = (__bf16)b.z; t[7] = (__bf16)b.w;
    *(bf16x8*)(out + idx) = t;
}

// ---------------------------------------------------------------------------
// Pre-pass: weights fp32 [k][n] -> bf16 transposed [n][k]
// ---------------------------------------------------------------------------
__global__ __launch_bounds__(256)
void wconv_kernel(const float* __restrict__ Wq, const float* __restrict__ Wk,
                  const float* __restrict__ Wv, const float* __restrict__ Wo,
                  __bf16* Wqt, __bf16* Wkt, __bf16* Wvt, __bf16* Wot) {
    const float* W; __bf16* T;
    switch (blockIdx.z) {
        case 0:  W = Wq; T = Wqt; break;
        case 1:  W = Wk; T = Wkt; break;
        case 2:  W = Wv; T = Wvt; break;
        default: W = Wo; T = Wot; break;
    }
    __shared__ float tile[32][33];
    const int n0 = blockIdx.x * 32, k0 = blockIdx.y * 32;
    const int tx = threadIdx.x & 31, ty = threadIdx.x >> 5;   // 32 x 8
#pragma unroll
    for (int i = 0; i < 4; ++i)
        tile[ty + 8 * i][tx] = W[(size_t)(k0 + ty + 8 * i) * DIM + n0 + tx];
    __syncthreads();
#pragma unroll
    for (int i = 0; i < 4; ++i)
        T[(size_t)(n0 + ty + 8 * i) * DIM + k0 + tx] = (__bf16)tile[tx][ty + 8 * i];
}

// ---------------------------------------------------------------------------
// GEMM: C[4096x1024] = A[4096x1024](bf16) @ W (Wt stored bf16 [n][k])
// block 256 (8 waves), tile 128x128, wave tile 64x32, K-slice 32,
// double-buffered async-to-LDS staging.
// MODE 0: fp32 row-major; MODE 1: bf16 [b][h][s][d]; MODE 2: bf16 [b][h][d][s]
// ---------------------------------------------------------------------------
static __device__ __forceinline__ void issue_slice(
        const __bf16* __restrict__ A, const __bf16* __restrict__ Wt,
        int bm, int bn, int kb, __bf16* AsBuf, __bf16* BsBuf, int tid) {
    // each tile: 128 rows x 32 bf16 (64B/row) = 8KB = 512 x 16B chunks
#pragma unroll
    for (int j = 0; j < 2; ++j) {
        const int c = j * 256 + tid;
        const int r = c >> 2, x = (c & 3) * 16;
        const char* g = (const char*)(A + (size_t)(bm + r) * DIM + kb) + x;
        async_b128((unsigned)(uintptr_t)((char*)AsBuf + c * 16), g);
    }
#pragma unroll
    for (int j = 0; j < 2; ++j) {
        const int c = j * 256 + tid;
        const int r = c >> 2, x = (c & 3) * 16;
        const char* g = (const char*)(Wt + (size_t)(bn + r) * DIM + kb) + x;
        async_b128((unsigned)(uintptr_t)((char*)BsBuf + c * 16), g);
    }
}

template <int MODE>
static __device__ void gemm_dev(const __bf16* __restrict__ A,
                                const __bf16* __restrict__ Wt, void* Cout) {
    __shared__ __bf16 As[2][128 * 32];
    __shared__ __bf16 Bs[2][128 * 32];

    const int tid = threadIdx.x;
    const int lane = tid & 31, w = tid >> 5;
    const int wm = w & 1, wn = w >> 1;               // wave grid 2(m) x 4(n)
    const int bm = blockIdx.y * 128, bn = blockIdx.x * 128;
    const int r15 = lane & 15, hi = lane >> 4;

    f32x8 acc[4][2];
#pragma unroll
    for (int mt = 0; mt < 4; ++mt)
#pragma unroll
        for (int nt = 0; nt < 2; ++nt) acc[mt][nt] = (f32x8)0.0f;

    issue_slice(A, Wt, bm, bn, 0, As[0], Bs[0], tid);

    for (int i = 0; i < DIM / 32; ++i) {
        const int cur = i & 1;
        if (i + 1 < DIM / 32) {
            issue_slice(A, Wt, bm, bn, (i + 1) * 32, As[cur ^ 1], Bs[cur ^ 1], tid);
            wait_async_4();   // in-order: first 4 (current slice) have landed
        } else {
            wait_async_0();
        }
        __syncthreads();

        bf16x16 af[4], bfr[2];
#pragma unroll
        for (int mt = 0; mt < 4; ++mt) {
            const __bf16* p = As[cur] + (wm * 64 + mt * 16 + r15) * 32 + hi * 8;
            af[mt] = mk16(*(const bf16x8*)p, *(const bf16x8*)(p + 16));
        }
#pragma unroll
        for (int nt = 0; nt < 2; ++nt) {
            const __bf16* p = Bs[cur] + (wn * 32 + nt * 16 + r15) * 32 + hi * 8;
            bfr[nt] = mk16(*(const bf16x8*)p, *(const bf16x8*)(p + 16));
        }
#pragma unroll
        for (int mt = 0; mt < 4; ++mt)
#pragma unroll
            for (int nt = 0; nt < 2; ++nt)
                acc[mt][nt] = wmma_bf16(af[mt], bfr[nt], acc[mt][nt]);
        __syncthreads();   // all waves done reading before this buffer is re-filled
    }

    // epilogue: C element e -> row = e + 8*(lane>=16), col = lane&15
#pragma unroll
    for (int mt = 0; mt < 4; ++mt)
#pragma unroll
        for (int nt = 0; nt < 2; ++nt)
#pragma unroll
            for (int e = 0; e < 8; ++e) {
                const int gm = bm + wm * 64 + mt * 16 + hi * 8 + e;
                const int gn = bn + wn * 32 + nt * 16 + r15;
                if constexpr (MODE == 0) {
                    ((float*)Cout)[(size_t)gm * DIM + gn] = acc[mt][nt][e];
                } else {
                    const int b = gm >> 11, s = gm & (SEQ - 1);
                    const int hh = gn >> 6, d = gn & (HDIM - 1);
                    size_t idx;
                    if constexpr (MODE == 1)
                        idx = ((size_t)(b * HEADS + hh) * SEQ + s) * HDIM + d;
                    else
                        idx = ((size_t)(b * HEADS + hh) * HDIM + d) * SEQ + s;
                    ((__bf16*)Cout)[idx] = (__bf16)acc[mt][nt][e];
                }
            }
}

__global__ __launch_bounds__(256)
void qkv_gemm_kernel(const __bf16* __restrict__ qb, const __bf16* __restrict__ kb,
                     const __bf16* __restrict__ vb, const __bf16* __restrict__ Wqt,
                     const __bf16* __restrict__ Wkt, const __bf16* __restrict__ Wvt,
                     __bf16* qw, __bf16* kw, __bf16* vw) {
    if (blockIdx.z == 0)      gemm_dev<1>(qb, Wqt, (void*)qw);
    else if (blockIdx.z == 1) gemm_dev<1>(kb, Wkt, (void*)kw);
    else                      gemm_dev<2>(vb, Wvt, (void*)vw);
}

__global__ __launch_bounds__(256)
void oproj_kernel(const __bf16* __restrict__ ctx, const __bf16* __restrict__ Wot,
                  float* out) {
    gemm_dev<0>(ctx, Wot, (void*)out);
}

// ---------------------------------------------------------------------------
// Flash attention: block = 128 threads (4 waves), 64 query rows per block,
// 64-key chunks, online softmax. q/k: [b][h][s][hd] bf16; v: [b][h][d][s].
// ---------------------------------------------------------------------------
__global__ __launch_bounds__(128)
void attn_kernel(const __bf16* __restrict__ qw, const __bf16* __restrict__ kw,
                 const __bf16* __restrict__ vw, const int* __restrict__ mask,
                 __bf16* __restrict__ ctx) {
    __shared__ __bf16 Ps[64 * 64];    // probabilities, C-layout -> A-layout bounce

    const int tid = threadIdx.x, lane = tid & 31, w = tid >> 5;
    const int bh = blockIdx.y, b = bh >> 4, h = bh & (HEADS - 1);
    const int qbase = blockIdx.x * 64;
    const int r15 = lane & 15, hi = lane >> 4;

    const __bf16* qptr = qw + (size_t)bh * SEQ * HDIM;
    const __bf16* kptr = kw + (size_t)bh * SEQ * HDIM;
    const __bf16* vptr = vw + (size_t)bh * HDIM * SEQ;

    // Q fragments (constant across key chunks): 2 k-steps over hd=64
    bf16x16 aq[2];
    {
        const __bf16* p0 = qptr + (size_t)(qbase + w * 16 + r15) * HDIM;
#pragma unroll
        for (int ks = 0; ks < 2; ++ks) {
            const __bf16* p = p0 + ks * 32 + hi * 8;
            aq[ks] = mk16(*(const bf16x8*)p, *(const bf16x8*)(p + 16));
        }
    }

    float mi[8], li[8];
    f32x8 oacc[4];
#pragma unroll
    for (int e = 0; e < 8; ++e) { mi[e] = -1e30f; li[e] = 0.0f; }
#pragma unroll
    for (int dt = 0; dt < 4; ++dt) oacc[dt] = (f32x8)0.0f;

    for (int kc = 0; kc < SEQ; kc += 64) {
        // ---- scores = Q @ K^T (B-frag = contiguous row of K[key][hd]) ----
        f32x8 sc[4];
#pragma unroll
        for (int nt = 0; nt < 4; ++nt) sc[nt] = (f32x8)0.0f;
#pragma unroll
        for (int ks = 0; ks < 2; ++ks)
#pragma unroll
            for (int nt = 0; nt < 4; ++nt) {
                const __bf16* p = kptr + (size_t)(kc + nt * 16 + r15) * HDIM
                                  + ks * 32 + hi * 8;
                bf16x16 bk = mk16(*(const bf16x8*)p, *(const bf16x8*)(p + 16));
                sc[nt] = wmma_bf16(aq[ks], bk, sc[nt]);
            }

        // ---- scale + mask ----
        float sv[4][8];
#pragma unroll
        for (int nt = 0; nt < 4; ++nt) {
            const int mv = mask[b * SEQ + kc + nt * 16 + r15];
            const float mbias = (mv == 0) ? -1e30f : 0.0f;
#pragma unroll
            for (int e = 0; e < 8; ++e) sv[nt][e] = sc[nt][e] * 0.125f + mbias;
        }

        // ---- online softmax (rows live in 16-lane halves) ----
#pragma unroll
        for (int e = 0; e < 8; ++e) {
            float cm = fmaxf(fmaxf(sv[0][e], sv[1][e]), fmaxf(sv[2][e], sv[3][e]));
            cm = halfmax16(cm);
            const float mnew = fmaxf(mi[e], cm);
            const float alpha = __expf(mi[e] - mnew);
            mi[e] = mnew;
            float rs = 0.0f;
#pragma unroll
            for (int nt = 0; nt < 4; ++nt) {
                const float p = __expf(sv[nt][e] - mnew);
                rs += p;
                Ps[(w * 16 + hi * 8 + e) * 64 + nt * 16 + r15] = (__bf16)p;
            }
            rs = halfsum16(rs);
            li[e] = li[e] * alpha + rs;
#pragma unroll
            for (int dt = 0; dt < 4; ++dt) oacc[dt][e] *= alpha;
        }
        __syncthreads();

        // ---- O += P @ V  (A-frag from Ps; B-frag contiguous from V^T) ----
#pragma unroll
        for (int ks = 0; ks < 2; ++ks) {
            const __bf16* pp = Ps + (w * 16 + r15) * 64 + ks * 32 + hi * 8;
            bf16x16 ap = mk16(*(const bf16x8*)pp, *(const bf16x8*)(pp + 16));
#pragma unroll
            for (int dt = 0; dt < 4; ++dt) {
                const __bf16* p = vptr + (size_t)(dt * 16 + r15) * SEQ
                                  + kc + ks * 32 + hi * 8;
                bf16x16 bv = mk16(*(const bf16x8*)p, *(const bf16x8*)(p + 16));
                oacc[dt] = wmma_bf16(ap, bv, oacc[dt]);
            }
        }
        __syncthreads();
    }

    // ---- normalize and scatter ctx as [b][s][h*64+d] bf16 ----
#pragma unroll
    for (int dt = 0; dt < 4; ++dt)
#pragma unroll
        for (int e = 0; e < 8; ++e) {
            const int row = qbase + w * 16 + hi * 8 + e;
            const int col = h * HDIM + dt * 16 + r15;
            ctx[(size_t)(b * SEQ + row) * DIM + col] = (__bf16)(oacc[dt][e] / li[e]);
        }
}

// ---------------------------------------------------------------------------
extern "C" void kernel_launch(void* const* d_in, const int* in_sizes, int n_in,
                              void* d_out, int out_size, void* d_ws, size_t ws_size,
                              hipStream_t stream) {
    (void)in_sizes; (void)n_in; (void)out_size; (void)ws_size;
    const float* q  = (const float*)d_in[0];
    const float* k  = (const float*)d_in[1];
    const float* v  = (const float*)d_in[2];
    const float* Wq = (const float*)d_in[3];
    const float* Wk = (const float*)d_in[4];
    const float* Wv = (const float*)d_in[5];
    const float* Wo = (const float*)d_in[6];
    const int* mask = (const int*)d_in[7];

    const size_t M4 = (size_t)MROWS * DIM;     // 4M elements
    const size_t M1 = (size_t)DIM * DIM;       // 1M elements
    __bf16* p   = (__bf16*)d_ws;
    __bf16* qb  = p;            p += M4;
    __bf16* kb  = p;            p += M4;
    __bf16* vb  = p;            p += M4;
    __bf16* Wqt = p;            p += M1;
    __bf16* Wkt = p;            p += M1;
    __bf16* Wvt = p;            p += M1;
    __bf16* Wot = p;            p += M1;
    __bf16* qw  = p;            p += M4;
    __bf16* kw  = p;            p += M4;
    __bf16* vw  = p;            p += M4;
    __bf16* ctx = p;            p += M4;
    float* out  = (float*)d_out;

    // 0) precision conversion pre-passes
    iconv_kernel<<<dim3((unsigned)(M4 / (256 * 8)), 1, 3), 256, 0, stream>>>(
        q, k, v, qb, kb, vb);
    wconv_kernel<<<dim3(DIM / 32, DIM / 32, 4), 256, 0, stream>>>(
        Wq, Wk, Wv, Wo, Wqt, Wkt, Wvt, Wot);

    // 1) fused QKV projections
    qkv_gemm_kernel<<<dim3(DIM / 128, MROWS / 128, 3), 256, 0, stream>>>(
        qb, kb, vb, Wqt, Wkt, Wvt, qw, kw, vw);

    // 2) flash attention
    attn_kernel<<<dim3(SEQ / 64, BATCH * HEADS), 128, 0, stream>>>(
        qw, kw, vw, mask, ctx);

    // 3) output projection
    oproj_kernel<<<dim3(DIM / 128, MROWS / 128), 256, 0, stream>>>(ctx, Wot, out);
}
